// Seq2SeqModel_2104533975552
// MI455X (gfx1250) — matrix operand order repaired and checked by reading, not compiled, not run on hardware
//
#include <hip/hip_runtime.h>

// ---------------------------------------------------------------------------
// Types / helpers
// ---------------------------------------------------------------------------
typedef __attribute__((ext_vector_type(16))) __bf16 v16bf;
typedef __attribute__((ext_vector_type(8)))  float  v8f;

__device__ __forceinline__ float bf2f(unsigned short h) {
  return __uint_as_float(((unsigned)h) << 16);
}
__device__ __forceinline__ unsigned short f2bf(float f) {
  unsigned u = __float_as_uint(f);
  u += 0x7fffu + ((u >> 16) & 1u);      // round-to-nearest-even
  return (unsigned short)(u >> 16);
}
__device__ __forceinline__ float sigf(float x) { return 1.0f / (1.0f + __expf(-x)); }

__device__ __forceinline__ v16bf frag2(uint4 a, uint4 b) {
  union { uint4 u[2]; v16bf v; } f;
  f.u[0] = a; f.u[1] = b;
  return f.v;
}

// ---------------------------------------------------------------------------
// WMMA GEMM:  C[M,N] = act( concat_K(A0,A1,A2)[M,K] * Bw[N,K]^T + bias[N] )
//   A segments row-major bf16 (lda = segment K), Bw row-major bf16 (ld = K).
//   Block 256 thr = 8 waves; wave tile 16x64 (4 wmma per K=32 chunk).
//   Requires: M%64==0, N%128==0, each segment K multiple of 32.
//   Segments expanded statically (macro) so every pointer derives directly
//   from a kernel argument -> compiler keeps global addrspace (global_load,
//   not flat_load) and the inner loop is pure pointer bumping.
// ---------------------------------------------------------------------------
__global__ __launch_bounds__(256) void wmma_gemm(
    const unsigned short* __restrict__ A0, int K0,
    const unsigned short* __restrict__ A1, int K1,
    const unsigned short* __restrict__ A2, int K2,
    const unsigned short* __restrict__ Bw,
    const float* __restrict__ bias,
    float* __restrict__ C, int ldc,
    unsigned short* __restrict__ Cbf, int ldcb,
    int act)
{
  const int lane = threadIdx.x & 31;
  const int w    = threadIdx.x >> 5;
  const int m0   = blockIdx.y * 64 + (w >> 1) * 16;
  const int n0   = blockIdx.x * 128 + (w & 1) * 64;
  const int lm   = lane & 15;   // A: row-in-tile, B/C: col-in-tile
  const int hi   = lane >> 4;   // lane half
  const int K    = K0 + K1 + K2;

  v8f acc[4] = {};

  // B fragment pointers: chunk k holds dwords [k/2 + hi*8 .. +7] of row n.
  const uint4* bp0 = (const uint4*)(Bw + (size_t)(n0 +  0 + lm) * (size_t)K) + 2 * hi;
  const uint4* bp1 = (const uint4*)(Bw + (size_t)(n0 + 16 + lm) * (size_t)K) + 2 * hi;
  const uint4* bp2 = (const uint4*)(Bw + (size_t)(n0 + 32 + lm) * (size_t)K) + 2 * hi;
  const uint4* bp3 = (const uint4*)(Bw + (size_t)(n0 + 48 + lm) * (size_t)K) + 2 * hi;

  // A fragment: chunk kl holds dwords [kl/2 + hi*8 .. +7].
#define GEMM_SEG(Aseg, Ksk)                                                     \
  if (Ksk) {                                                                    \
    const uint4* ap = (const uint4*)(Aseg + (size_t)(m0 + lm) * (size_t)Ksk) + 2 * hi; \
    _Pragma("unroll 1")                                                         \
    for (int k = 0; k < Ksk; k += 32) {                                         \
      uint4 a0 = ap[0], a1 = ap[1];                                             \
      uint4 c00 = bp0[0], c01 = bp0[1];                                         \
      uint4 c10 = bp1[0], c11 = bp1[1];                                         \
      uint4 c20 = bp2[0], c21 = bp2[1];                                         \
      uint4 c30 = bp3[0], c31 = bp3[1];                                         \
      ap += 4; bp0 += 4; bp1 += 4; bp2 += 4; bp3 += 4;                          \
      v16bf afrag = frag2(a0, a1);                                              \
      acc[0] = __builtin_amdgcn_wmma_f32_16x16x32_bf16(                         \
          false, afrag, false, frag2(c00, c01), (short)0, acc[0], false, false);\
      acc[1] = __builtin_amdgcn_wmma_f32_16x16x32_bf16(                         \
          false, afrag, false, frag2(c10, c11), (short)0, acc[1], false, false);\
      acc[2] = __builtin_amdgcn_wmma_f32_16x16x32_bf16(                         \
          false, afrag, false, frag2(c20, c21), (short)0, acc[2], false, false);\
      acc[3] = __builtin_amdgcn_wmma_f32_16x16x32_bf16(                         \
          false, afrag, false, frag2(c30, c31), (short)0, acc[3], false, false);\
    }                                                                           \
  }

  GEMM_SEG(A0, K0)
  GEMM_SEG(A1, K1)
  GEMM_SEG(A2, K2)
#undef GEMM_SEG

#pragma unroll
  for (int t = 0; t < 4; ++t) {
    const int n = n0 + t * 16 + lm;
    const float bv = bias ? bias[n] : 0.0f;
#pragma unroll
    for (int r = 0; r < 8; ++r) {
      float v = acc[t][r] + bv;
      if (act) v = tanhf(v);
      const int m = m0 + r + hi * 8;
      if (C)   C[(size_t)m * ldc + n] = v;
      if (Cbf) Cbf[(size_t)m * ldcb + n] = f2bf(v);
    }
  }
}

// ---------------------------------------------------------------------------
// f32 -> bf16 weight convert with source slicing (for Wa[:, :H] / Wa[:, H:])
// ---------------------------------------------------------------------------
__global__ void cvt_w(const float* __restrict__ src, unsigned short* __restrict__ dst,
                      int n, int cols, int lds, int soff)
{
  int i = blockIdx.x * 256 + threadIdx.x;
  if (i >= n) return;
  int r = i / cols, c = i - r * cols;
  dst[i] = f2bf(src[(size_t)r * lds + soff + c]);
}

// x_all[s,b,:] = bf16(enc_emb[src[b,s],:])   layout (S,B,E)
__global__ void embed_enc(const int* __restrict__ src, const unsigned short* __restrict__ emb,
                          unsigned short* __restrict__ x_all, int S, int B, int E)
{
  long long i = (long long)blockIdx.x * 256 + threadIdx.x;
  long long total = (long long)S * B * E;
  if (i >= total) return;
  int e = (int)(i % E);
  long long sb = i / E;
  int b = (int)(sb % B);
  int s = (int)(sb / B);
  x_all[i] = emb[(size_t)src[(size_t)b * S + s] * E + e];
}

// GRU gate fusion: h2 = (1-z)*n + z*h ; writes f32 h, bf16 h, optional strided bf16
__global__ void gru_gate(const float* __restrict__ gi, const float* __restrict__ gh,
                         const float* __restrict__ hprev,
                         float* __restrict__ hnew, unsigned short* __restrict__ hbf,
                         unsigned short* __restrict__ outs, int ostride,
                         int B, int H)
{
  int i = blockIdx.x * 256 + threadIdx.x;
  if (i >= B * H) return;
  int b = i / H, j = i - b * H;
  size_t gb = (size_t)b * 3 * H;
  float r = sigf(gi[gb + j] + gh[gb + j]);
  float z = sigf(gi[gb + H + j] + gh[gb + H + j]);
  float n = tanhf(gi[gb + 2 * H + j] + r * gh[gb + 2 * H + j]);
  float h2 = (1.0f - z) * n + z * hprev[i];
  hnew[i] = h2;
  unsigned short h16 = f2bf(h2);
  hbf[i] = h16;
  if (outs) outs[(size_t)b * ostride + j] = h16;
}

// cond = bf16(concat(hf, hb, mode_E[m], tense_E[t], person_E[p]))  (B, 2H+3C)
__global__ void build_cond(const float* __restrict__ hf, const float* __restrict__ hb,
                           const int* __restrict__ mi, const int* __restrict__ ti,
                           const int* __restrict__ pi,
                           const float* __restrict__ mE, const float* __restrict__ tE,
                           const float* __restrict__ pE,
                           unsigned short* __restrict__ cond, int B, int H, int C)
{
  int D = 2 * H + 3 * C;
  int i = blockIdx.x * 256 + threadIdx.x;
  if (i >= B * D) return;
  int b = i / D, d = i - b * D;
  float v;
  if (d < H)            v = hf[(size_t)b * H + d];
  else if (d < 2 * H)   v = hb[(size_t)b * H + d - H];
  else {
    int dd = d - 2 * H;
    if (dd < C)          v = mE[(size_t)mi[b] * C + dd];
    else if (dd < 2 * C) v = tE[(size_t)ti[b] * C + dd - C];
    else                 v = pE[(size_t)pi[b] * C + dd - 2 * C];
  }
  cond[i] = f2bf(v);
}

// scores[b,s] = sum_h va[h]*tanh(hWa[b,h] + enc_proj[b,s,h]); mask; softmax -> wat
__global__ void attn_scores(const float* __restrict__ hWa,
                            const unsigned short* __restrict__ enc_proj,
                            const float* __restrict__ va, const int* __restrict__ src,
                            float* __restrict__ wat, int S, int H)
{
  const int b = blockIdx.x;
  __shared__ float sc[128];
  const int lane = threadIdx.x & 31;
  const int wv   = threadIdx.x >> 5;          // 8 waves
  for (int s = wv; s < S; s += 8) {
    float p = 0.0f;
    const unsigned short* ep = enc_proj + ((size_t)b * S + s) * H;
    for (int h = lane; h < H; h += 32)
      p += va[h] * tanhf(hWa[(size_t)b * H + h] + bf2f(ep[h]));
    for (int off = 16; off; off >>= 1) p += __shfl_xor(p, off, 32);
    if (lane == 0) sc[s] = (src[(size_t)b * S + s] != 0) ? p : -1e9f;
  }
  __syncthreads();
  if (threadIdx.x == 0) {
    float mx = -1e30f;
    for (int s = 0; s < S; ++s) mx = fmaxf(mx, sc[s]);
    float sum = 0.0f;
    for (int s = 0; s < S; ++s) { float e = expf(sc[s] - mx); sc[s] = e; sum += e; }
    float inv = 1.0f / sum;
    for (int s = 0; s < S; ++s) wat[(size_t)b * S + s] = sc[s] * inv;
  }
}

// ctx[b,d] = sum_s wat[b,s] * enc_out[b,s,d]
__global__ void ctx_kernel(const float* __restrict__ wat,
                           const unsigned short* __restrict__ enc_out,
                           float* __restrict__ ctx, unsigned short* __restrict__ ctxbf,
                           int B, int S, int D2)
{
  int i = blockIdx.x * 256 + threadIdx.x;
  if (i >= B * D2) return;
  int b = i / D2, d = i - b * D2;
  const unsigned short* base = enc_out + (size_t)b * S * D2 + d;
  const float* wb = wat + (size_t)b * S;
  float acc = 0.0f;
  for (int s = 0; s < S; ++s) acc += wb[s] * bf2f(base[(size_t)s * D2]);
  ctx[i] = acc;
  ctxbf[i] = f2bf(acc);
}

// per-b argmax over V=256 then gather dec embedding (bf16)
__global__ void argmax_embed(const float* __restrict__ pred, int ldp,
                             const unsigned short* __restrict__ demb,
                             unsigned short* __restrict__ embout, int E)
{
  const int b = blockIdx.x;
  __shared__ float sv[256];
  __shared__ int   si[256];
  int t = threadIdx.x;
  sv[t] = pred[(size_t)b * ldp + t];
  si[t] = t;
  __syncthreads();
  for (int off = 128; off; off >>= 1) {
    if (t < off) {
      if (sv[t + off] > sv[t] || (sv[t + off] == sv[t] && si[t + off] < si[t])) {
        sv[t] = sv[t + off]; si[t] = si[t + off];
      }
    }
    __syncthreads();
  }
  int tok = si[0];
  for (int e = t; e < E; e += 256)
    embout[(size_t)b * E + e] = demb[(size_t)tok * E + e];
}

// first decoder token: emb = dec_emb[tgt[:,0]]
__global__ void init_embed(const int* __restrict__ tgt, int T,
                           const unsigned short* __restrict__ demb,
                           unsigned short* __restrict__ embout, int B, int E)
{
  int i = blockIdx.x * 256 + threadIdx.x;
  if (i >= B * E) return;
  int b = i / E, e = i - b * E;
  embout[i] = demb[(size_t)tgt[(size_t)b * T] * E + e];
}

// ---------------------------------------------------------------------------
// Host
// ---------------------------------------------------------------------------
extern "C" void kernel_launch(void* const* d_in, const int* in_sizes, int n_in,
                              void* d_out, int out_size, void* d_ws, size_t ws_size,
                              hipStream_t stream) {
  const int V = 256, E = 512, H = 1024, C = 128;
  const int B = 256, S = 96, T = 40;
  const int H3 = 3 * H, H2 = 2 * H;
  const int Dc = 2 * H + 3 * C;      // 2432
  const int Kdec = E + 2 * H;        // 2560
  const int Kfc = H + 2 * H + E;     // 3584

  const int*   src      = (const int*)d_in[0];
  const int*   tgt      = (const int*)d_in[1];
  const int*   mode_i   = (const int*)d_in[2];
  const int*   tense_i  = (const int*)d_in[3];
  const int*   person_i = (const int*)d_in[4];
  const float* enc_emb  = (const float*)d_in[5];
  const float* eWihF    = (const float*)d_in[6];
  const float* eWhhF    = (const float*)d_in[7];
  const float* ebihF    = (const float*)d_in[8];
  const float* ebhhF    = (const float*)d_in[9];
  const float* eWihB    = (const float*)d_in[10];
  const float* eWhhB    = (const float*)d_in[11];
  const float* ebihB    = (const float*)d_in[12];
  const float* ebhhB    = (const float*)d_in[13];
  const float* Wa       = (const float*)d_in[14];
  const float* va       = (const float*)d_in[15];
  const float* dec_emb  = (const float*)d_in[16];
  const float* dWih     = (const float*)d_in[17];
  const float* dWhh     = (const float*)d_in[18];
  const float* dbih     = (const float*)d_in[19];
  const float* dbhh     = (const float*)d_in[20];
  const float* Wfc      = (const float*)d_in[21];
  const float* bfc      = (const float*)d_in[22];
  const float* modeE    = (const float*)d_in[23];
  const float* tenseE   = (const float*)d_in[24];
  const float* personE  = (const float*)d_in[25];
  const float* Wbr      = (const float*)d_in[26];
  const float* bbr      = (const float*)d_in[27];

  // workspace allocator
  char* wsp = (char*)d_ws;
  size_t off = 0;
  auto alloc = [&](size_t bytes) -> void* {
    void* p = wsp + off;
    off = (off + bytes + 255) & ~(size_t)255;
    return p;
  };
  typedef unsigned short u16;
  u16* eembbf  = (u16*)alloc((size_t)V * E * 2);
  u16* dembbf  = (u16*)alloc((size_t)V * E * 2);
  u16* wihf    = (u16*)alloc((size_t)H3 * E * 2);
  u16* whhf    = (u16*)alloc((size_t)H3 * H * 2);
  u16* wihb    = (u16*)alloc((size_t)H3 * E * 2);
  u16* whhb    = (u16*)alloc((size_t)H3 * H * 2);
  u16* wah     = (u16*)alloc((size_t)H * H * 2);
  u16* wae     = (u16*)alloc((size_t)H * H2 * 2);
  u16* dwih    = (u16*)alloc((size_t)H3 * Kdec * 2);
  u16* dwhh    = (u16*)alloc((size_t)H3 * H * 2);
  u16* wfcb    = (u16*)alloc((size_t)V * Kfc * 2);
  u16* wbrb    = (u16*)alloc((size_t)H * Dc * 2);
  u16* x_all   = (u16*)alloc((size_t)S * B * E * 2);
  u16* encout  = (u16*)alloc((size_t)B * S * H2 * 2);
  u16* encproj = (u16*)alloc((size_t)B * S * H * 2);
  float* hF    = (float*)alloc((size_t)B * H * 4);
  float* hB    = (float*)alloc((size_t)B * H * 4);
  float* hD    = (float*)alloc((size_t)B * H * 4);
  u16* hFbf    = (u16*)alloc((size_t)B * H * 2);
  u16* hBbf    = (u16*)alloc((size_t)B * H * 2);
  u16* hDbf    = (u16*)alloc((size_t)B * H * 2);
  float* gi    = (float*)alloc((size_t)B * H3 * 4);
  float* gh    = (float*)alloc((size_t)B * H3 * 4);
  float* hWa   = (float*)alloc((size_t)B * H * 4);
  float* wat   = (float*)alloc((size_t)B * S * 4);
  float* ctx   = (float*)alloc((size_t)B * H2 * 4);
  u16* ctxbf   = (u16*)alloc((size_t)B * H2 * 2);
  u16* embbf   = (u16*)alloc((size_t)B * E * 2);
  u16* condbf  = (u16*)alloc((size_t)B * Dc * 2);
  (void)ws_size; (void)in_sizes; (void)n_in; (void)out_size;

  auto cvt = [&](const float* s, u16* dst, int rows, int cols, int lds, int soff) {
    int n = rows * cols;
    cvt_w<<<(n + 255) / 256, 256, 0, stream>>>(s, dst, n, cols, lds, soff);
  };
  auto gemm = [&](const u16* A0, int K0, const u16* A1, int K1, const u16* A2, int K2,
                  const u16* Bw, const float* bias, float* Cf, int ldc,
                  u16* Cbf, int ldcb, int M, int N, int act) {
    dim3 g(N / 128, M / 64);
    wmma_gemm<<<g, 256, 0, stream>>>(A0, K0, A1, K1, A2, K2, Bw, bias,
                                     Cf, ldc, Cbf, ldcb, act);
  };

  // ---- weight / embedding conversion (bf16) ----
  cvt(enc_emb, eembbf, V, E, E, 0);
  cvt(dec_emb, dembbf, V, E, E, 0);
  cvt(eWihF, wihf, H3, E, E, 0);
  cvt(eWhhF, whhf, H3, H, H, 0);
  cvt(eWihB, wihb, H3, E, E, 0);
  cvt(eWhhB, whhb, H3, H, H, 0);
  cvt(Wa, wah, H, H, 3 * H, 0);        // Wa[:, :H]
  cvt(Wa, wae, H, H2, 3 * H, H);       // Wa[:, H:]
  cvt(dWih, dwih, H3, Kdec, Kdec, 0);
  cvt(dWhh, dwhh, H3, H, H, 0);
  cvt(Wfc, wfcb, V, Kfc, Kfc, 0);
  cvt(Wbr, wbrb, H, Dc, Dc, 0);

  // ---- encoder embedding gather ----
  {
    long long n = (long long)S * B * E;
    embed_enc<<<(unsigned)((n + 255) / 256), 256, 0, stream>>>(src, eembbf, x_all, S, B, E);
  }

  // ---- encoder, forward direction ----
  hipMemsetAsync(hF, 0, (size_t)B * H * 4, stream);
  hipMemsetAsync(hFbf, 0, (size_t)B * H * 2, stream);
  const int BH = B * H;
  for (int s = 0; s < S; ++s) {
    gemm(x_all + (size_t)s * B * E, E, nullptr, 0, nullptr, 0,
         wihf, ebihF, gi, H3, nullptr, 0, B, H3, 0);
    gemm(hFbf, H, nullptr, 0, nullptr, 0,
         whhf, ebhhF, gh, H3, nullptr, 0, B, H3, 0);
    gru_gate<<<(BH + 255) / 256, 256, 0, stream>>>(
        gi, gh, hF, hF, hFbf, encout + (size_t)s * H2, S * H2, B, H);
  }
  // ---- encoder, backward direction ----
  hipMemsetAsync(hB, 0, (size_t)B * H * 4, stream);
  hipMemsetAsync(hBbf, 0, (size_t)B * H * 2, stream);
  for (int s = S - 1; s >= 0; --s) {
    gemm(x_all + (size_t)s * B * E, E, nullptr, 0, nullptr, 0,
         wihb, ebihB, gi, H3, nullptr, 0, B, H3, 0);
    gemm(hBbf, H, nullptr, 0, nullptr, 0,
         whhb, ebhhB, gh, H3, nullptr, 0, B, H3, 0);
    gru_gate<<<(BH + 255) / 256, 256, 0, stream>>>(
        gi, gh, hB, hB, hBbf, encout + (size_t)s * H2 + H, S * H2, B, H);
  }

  // ---- attention pre-projection: enc_proj = enc_out @ Wa_e.T  (bf16 out only) ----
  gemm(encout, H2, nullptr, 0, nullptr, 0,
       wae, nullptr, nullptr, 0, encproj, H, B * S, H, 0);

  // ---- decoder init: cond -> dec_h0 = tanh(cond @ Wbr.T + bbr) ----
  {
    int n = B * Dc;
    build_cond<<<(n + 255) / 256, 256, 0, stream>>>(
        hF, hB, mode_i, tense_i, person_i, modeE, tenseE, personE, condbf, B, H, C);
  }
  gemm(condbf, Dc, nullptr, 0, nullptr, 0,
       wbrb, bbr, hD, H, hDbf, H, B, H, 1);
  {
    int n = B * E;
    init_embed<<<(n + 255) / 256, 256, 0, stream>>>(tgt, T, dembbf, embbf, B, E);
  }

  // ---- decoder loop (greedy) ----
  float* out = (float*)d_out;
  const int ldo = (T - 1) * V;
  for (int t = 0; t < T - 1; ++t) {
    // attention
    gemm(hDbf, H, nullptr, 0, nullptr, 0,
         wah, nullptr, hWa, H, nullptr, 0, B, H, 0);
    attn_scores<<<B, 256, 0, stream>>>(hWa, encproj, va, src, wat, S, H);
    {
      int n = B * H2;
      ctx_kernel<<<(n + 255) / 256, 256, 0, stream>>>(wat, encout, ctx, ctxbf, B, S, H2);
    }
    // GRU step: rnn_in = [emb | ctx]
    gemm(embbf, E, ctxbf, H2, nullptr, 0,
         dwih, dbih, gi, H3, nullptr, 0, B, H3, 0);
    gemm(hDbf, H, nullptr, 0, nullptr, 0,
         dwhh, dbhh, gh, H3, nullptr, 0, B, H3, 0);
    gru_gate<<<(BH + 255) / 256, 256, 0, stream>>>(
        gi, gh, hD, hD, hDbf, nullptr, 0, B, H);
    // pred = [h2 | ctx | emb] @ Wfc.T + bfc  -> directly into d_out[:, t, :]
    gemm(hDbf, H, ctxbf, H2, embbf, E,
         wfcb, bfc, out + (size_t)t * V, ldo, nullptr, 0, B, V, 0);
    // next token
    if (t < T - 2)
      argmax_embed<<<B, 256, 0, stream>>>(out + (size_t)t * V, ldo, dembbf, embbf, E);
  }
}
